// DirectContractedVoxGO_36481452212865
// MI455X (gfx1250) — matrix operand order repaired and checked by reading, not compiled, not run on hardware
//
#include <hip/hip_runtime.h>
#include <hip/hip_bf16.h>
#include <cstdint>

typedef __attribute__((ext_vector_type(16))) _Float16 v16h;
typedef __attribute__((ext_vector_type(8)))  _Float16 v8h;
typedef __attribute__((ext_vector_type(8)))  float    v8f;

#define N_RAYS_   4096
#define WORLD_R   160
#define K0_DIM_   12
#define N_INNER_  231
#define N_OUTER_  23
#define P_        254                 /* samples per ray (stepsize=1) */
#define M_        (N_RAYS_ * P_)      /* 1,040,384 points, divisible by 128 */
#define BG_LEN_   0.2f
#define XYZ_MIN_  (-1.2f)
#define XYZ_RNG_  2.4f
#define ACT_SHIFT_ (-9.21024036697585f)
#define SQRT3_    1.7320508075688772f
#define BG_       1.0f

// LDS halves: w0T [128][64] + w1T [128][128] + w2T [16][128] + stage [8][16][128]
#define W0_HALVES   (128*64)
#define W1_HALVES   (128*128)
#define W2_HALVES   (16*128)
#define WTOT_HALVES (W0_HALVES + W1_HALVES + W2_HALVES)   /* 26624 */
#define STAGE_HALVES (8*16*128)                            /* 16384 */
#define SMEM_BYTES  ((WTOT_HALVES + STAGE_HALVES) * 2)     /* 86016 */

// ---------------------------------------------------------------- geometry

__device__ __forceinline__ void ray_point(const float* __restrict__ ro,
                                          const float* __restrict__ rd,
                                          int p, float& px, float& py, float& pz)
{
    int ray = p / P_;
    int j   = p - ray * P_;
    float ox = ro[ray*3+0], oy = ro[ray*3+1], oz = ro[ray*3+2];
    float dx = rd[ray*3+0], dy = rd[ray*3+1], dz = rd[ray*3+2];
    float inv = rsqrtf(dx*dx + dy*dy + dz*dz);
    dx *= inv; dy *= inv; dz *= inv;
    float t;
    if (j < N_INNER_) {
        t = ((float)j + 0.5f) * (2.0f * SQRT3_ / (float)N_INNER_);
    } else {
        int i = j - N_INNER_;
        float step = 0.999f / (float)N_OUTER_;
        float d0 = 1.0f - (float)i     * step;
        float d1 = 1.0f - (float)(i+1) * step;
        t = SQRT3_ * (1.0f/d0 + 1.0f/d1);
    }
    px = ox + dx*t; py = oy + dy*t; pz = oz + dz*t;
    float nrm = fmaxf(fabsf(px), fmaxf(fabsf(py), fabsf(pz)));
    if (nrm > 1.0f) {
        float safe = fmaxf(nrm, 1e-9f);
        float s = (1.0f + BG_LEN_ - BG_LEN_/safe) / safe;
        px *= s; py *= s; pz *= s;
    }
}

__device__ __forceinline__ void grid_setup(float px, float py, float pz,
                                           int& ix, int& iy, int& iz,
                                           float& fx, float& fy, float& fz)
{
    const float Rm1 = (float)(WORLD_R - 1);
    float gx = fminf(fmaxf((px - XYZ_MIN_) * (Rm1 / XYZ_RNG_), 0.0f), Rm1);
    float gy = fminf(fmaxf((py - XYZ_MIN_) * (Rm1 / XYZ_RNG_), 0.0f), Rm1);
    float gz = fminf(fmaxf((pz - XYZ_MIN_) * (Rm1 / XYZ_RNG_), 0.0f), Rm1);
    ix = (int)floorf(gx); if (ix > WORLD_R-2) ix = WORLD_R-2;
    iy = (int)floorf(gy); if (iy > WORLD_R-2) iy = WORLD_R-2;
    iz = (int)floorf(gz); if (iz > WORLD_R-2) iz = WORLD_R-2;
    fx = gx - (float)ix; fy = gy - (float)iy; fz = gz - (float)iz;
}

__device__ __forceinline__ float tri(const float* __restrict__ g,
                                     int ix, int iy, int iz,
                                     float fx, float fy, float fz)
{
    const int R = WORLD_R;
    const float* b0 = g + ((size_t)ix * R + iy) * R + iz;
    const float* b1 = b0 + (size_t)R * R;
    float c000 = b0[0],   c001 = b0[1];
    float c010 = b0[R],   c011 = b0[R+1];
    float c100 = b1[0],   c101 = b1[1];
    float c110 = b1[R],   c111 = b1[R+1];
    float c00 = c000*(1.f-fz) + c001*fz;
    float c01 = c010*(1.f-fz) + c011*fz;
    float c10 = c100*(1.f-fz) + c101*fz;
    float c11 = c110*(1.f-fz) + c111*fz;
    float c0 = c00*(1.f-fy) + c01*fy;
    float c1 = c10*(1.f-fy) + c11*fy;
    return c0*(1.f-fx) + c1*fx;
}

// ---------------------------------------------------------------- WMMA fragment loads
// A (16x32 f16): lane<16 row M=lane holds K {kb..kb+7, kb+16..kb+23} with kb=0;
//                lane>=16 same row, kb=8.  (ISA 7.12.2)
__device__ __forceinline__ v16h load_afrag(const _Float16* row, int kc, int lane)
{
    int kb = kc*32 + ((lane < 16) ? 0 : 8);
    v8h lo = *(const v8h*)(row + kb);
    v8h hi = *(const v8h*)(row + kb + 16);
    return __builtin_shufflevector(lo, hi, 0,1,2,3,4,5,6,7,8,9,10,11,12,13,14,15);
}

// B (32x16 f16): lane holds column N = lane&15; lane<16 -> K {0..7,16..23}, lane>=16 -> K {8..15,24..31}
// wT is [N][K] row-major f16 (K contiguous) -> two 16B LDS loads.
__device__ __forceinline__ v16h load_bfrag(const _Float16* wT,
                                           int ldk, int nt, int kc, int lane)
{
    int n  = nt*16 + (lane & 15);
    int kb = kc*32 + ((lane < 16) ? 0 : 8);
    const _Float16* p = wT + (size_t)n * ldk + kb;
    v8h lo = *(const v8h*)(p);
    v8h hi = *(const v8h*)(p + 16);
    return __builtin_shufflevector(lo, hi, 0,1,2,3,4,5,6,7,8,9,10,11,12,13,14,15);
}

// ---------------------------------------------------------------- kernels

__global__ void prep_weights_kernel(const float* __restrict__ w0,
                                    const float* __restrict__ w1,
                                    const float* __restrict__ w2,
                                    _Float16* __restrict__ wAll)  // w0T|w1T|w2T contiguous
{
    int i = blockIdx.x * blockDim.x + threadIdx.x;
    if (i < W0_HALVES) {
        int n = i / 64, k = i % 64;
        wAll[i] = (_Float16)((k < 39) ? w0[k*128 + n] : 0.0f);
    } else if (i < W0_HALVES + W1_HALVES) {
        int j = i - W0_HALVES;
        int n = j / 128, k = j % 128;
        wAll[i] = (_Float16)w1[k*128 + n];
    } else if (i < WTOT_HALVES) {
        int j = i - (W0_HALVES + W1_HALVES);
        int n = j / 128, k = j % 128;
        wAll[i] = (_Float16)((n < 3) ? w2[k*3 + n] : 0.0f);
    }
}

__global__ void alpha_kernel(const float* __restrict__ ro,
                             const float* __restrict__ rd,
                             const float* __restrict__ dgrid,
                             const int*   __restrict__ stepsize,
                             float* __restrict__ alpha)
{
    int p = blockIdx.x * blockDim.x + threadIdx.x;
    if (p >= M_) return;
    float px, py, pz;
    ray_point(ro, rd, p, px, py, pz);
    int ix, iy, iz; float fx, fy, fz;
    grid_setup(px, py, pz, ix, iy, iz, fx, fy, fz);
    float d = tri(dgrid, ix, iy, iz, fx, fy, fz);
    float interval = (float)(*stepsize);
    float a = 1.0f - __powf(1.0f + __expf(d + ACT_SHIFT_), -interval);
    alpha[p] = a;
}

__global__ void scan_kernel(const float* __restrict__ alpha,
                            float* __restrict__ wgt,
                            float* __restrict__ alast)
{
    int ray = blockIdx.x * blockDim.x + threadIdx.x;
    if (ray >= N_RAYS_) return;
    float T = 1.0f;
    const float* a = alpha + (size_t)ray * P_;
    float* w = wgt + (size_t)ray * P_;
    for (int j = 0; j < P_; ++j) {
        float av = a[j];
        w[j] = av * T;
        T *= (1.0f - av);
    }
    alast[ray] = T;
}

__global__ __launch_bounds__(256)
void fused_mlp_kernel(const float* __restrict__ ro,
                      const float* __restrict__ rdir,
                      const float* __restrict__ vdirs,
                      const float* __restrict__ k0g,
                      const _Float16* __restrict__ wAll,
                      const float* __restrict__ b0,
                      const float* __restrict__ b1,
                      const float* __restrict__ b2,
                      const float* __restrict__ wgt,
                      float* __restrict__ rgbw)
{
    extern __shared__ __align__(16) char smem[];
    _Float16* wlds = (_Float16*)smem;
    const _Float16* w0L = wlds;
    const _Float16* w1L = wlds + W0_HALVES;
    const _Float16* w2L = wlds + W0_HALVES + W1_HALVES;
    _Float16* stage = wlds + WTOT_HALVES;   // [8 waves][16 rows][128 cols]

    const int lane = threadIdx.x & 31;
    const int wave = threadIdx.x >> 5;
    const int ncol = lane & 15;
    const int mofs = (lane < 16) ? 0 : 8;
    const int NTILES = M_ / 128;

    // cooperative one-shot weight load into LDS (w0T|w1T|w2T are contiguous in ws)
    {
        const int nvec = WTOT_HALVES / 8;   // 16B chunks
        const int4* src = (const int4*)wAll;
        int4* dst = (int4*)wlds;
        for (int i = threadIdx.x; i < nvec; i += blockDim.x) dst[i] = src[i];
    }
    __syncthreads();

    for (int tile = blockIdx.x; tile < NTILES; tile += gridDim.x) {
        // LDS fence each iteration: keeps B-fragment ds_loads inside the loop
        // (prevents LLVM hoisting 52 fragments -> scratch spills)
        __syncthreads();

        const int base = tile * 128 + wave * 16;
        _Float16* wstage = stage + (size_t)wave * 16 * 128;

        // ---- features: lanes 0..15 each build one row [k0(12)|vd(3)|sin(12)|cos(12)|0pad]
        if (lane < 16) {
            int p = base + lane;
            float px, py, pz;
            ray_point(ro, rdir, p, px, py, pz);
            int ix, iy, iz; float fx, fy, fz;
            grid_setup(px, py, pz, ix, iy, iz, fx, fy, fz);
            _Float16* row = wstage + (size_t)lane * 128;
            const size_t R3 = (size_t)WORLD_R * WORLD_R * WORLD_R;
            #pragma unroll
            for (int c = 0; c < K0_DIM_; ++c)
                row[c] = (_Float16)tri(k0g + (size_t)c * R3, ix, iy, iz, fx, fy, fz);
            int ray = p / P_;
            float vx = vdirs[ray*3+0], vy = vdirs[ray*3+1], vz = vdirs[ray*3+2];
            row[12] = (_Float16)vx; row[13] = (_Float16)vy; row[14] = (_Float16)vz;
            float v3[3] = {vx, vy, vz};
            #pragma unroll
            for (int d = 0; d < 3; ++d) {
                float f = 1.0f;
                #pragma unroll
                for (int q = 0; q < 4; ++q) {
                    float ve = v3[d] * f;
                    row[15 + d*4 + q] = (_Float16)__sinf(ve);
                    row[27 + d*4 + q] = (_Float16)__cosf(ve);
                    f *= 2.0f;
                }
            }
            #pragma unroll
            for (int c = 39; c < 64; ++c) row[c] = (_Float16)0.0f;
        }

        const _Float16* arow = wstage + (size_t)ncol * 128;

        // ---- layer 0: [16x64] x [64x128]
        v16h a0[2];
        #pragma unroll
        for (int kc = 0; kc < 2; ++kc) a0[kc] = load_afrag(arow, kc, lane);
        #pragma unroll
        for (int nt = 0; nt < 8; ++nt) {
            v8f acc = {0.f,0.f,0.f,0.f,0.f,0.f,0.f,0.f};
            #pragma unroll
            for (int kc = 0; kc < 2; ++kc) {
                v16h b = load_bfrag(w0L, 64, nt, kc, lane);
                acc = __builtin_amdgcn_wmma_f32_16x16x32_f16(
                        false, a0[kc], false, b, (short)0, acc, false, false);
            }
            float bias = b0[nt*16 + ncol];
            #pragma unroll
            for (int r = 0; r < 8; ++r) {
                float v = acc[r] + bias;
                v = v > 0.0f ? v : 0.0f;
                wstage[(size_t)(r + mofs) * 128 + nt*16 + ncol] = (_Float16)v;
            }
        }

        // ---- layer 1: [16x128] x [128x128]
        v16h a1[4];
        #pragma unroll
        for (int kc = 0; kc < 4; ++kc) a1[kc] = load_afrag(arow, kc, lane);
        #pragma unroll
        for (int nt = 0; nt < 8; ++nt) {
            v8f acc = {0.f,0.f,0.f,0.f,0.f,0.f,0.f,0.f};
            #pragma unroll
            for (int kc = 0; kc < 4; ++kc) {
                v16h b = load_bfrag(w1L, 128, nt, kc, lane);
                acc = __builtin_amdgcn_wmma_f32_16x16x32_f16(
                        false, a1[kc], false, b, (short)0, acc, false, false);
            }
            float bias = b1[nt*16 + ncol];
            #pragma unroll
            for (int r = 0; r < 8; ++r) {
                float v = acc[r] + bias;
                v = v > 0.0f ? v : 0.0f;
                wstage[(size_t)(r + mofs) * 128 + nt*16 + ncol] = (_Float16)v;
            }
        }

        // ---- layer 2: [16x128] x [128x16(pad)], only N=0..2 valid
        v16h a2[4];
        #pragma unroll
        for (int kc = 0; kc < 4; ++kc) a2[kc] = load_afrag(arow, kc, lane);
        v8f acc2 = {0.f,0.f,0.f,0.f,0.f,0.f,0.f,0.f};
        #pragma unroll
        for (int kc = 0; kc < 4; ++kc) {
            v16h b = load_bfrag(w2L, 128, 0, kc, lane);
            acc2 = __builtin_amdgcn_wmma_f32_16x16x32_f16(
                     false, a2[kc], false, b, (short)0, acc2, false, false);
        }
        if (ncol < 3) {
            float bias = b2[ncol];
            #pragma unroll
            for (int r = 0; r < 8; ++r) {
                int p = base + r + mofs;
                float v = acc2[r] + bias;
                v = 1.0f / (1.0f + __expf(-v));           // sigmoid
                rgbw[(size_t)p * 3 + ncol] = wgt[p] * v;  // deterministic: summed in order later
            }
        }
    }
}

__global__ void reduce_rays_kernel(const float* __restrict__ rgbw,
                                   const float* __restrict__ alast,
                                   float* __restrict__ out)
{
    int idx = blockIdx.x * blockDim.x + threadIdx.x;
    if (idx >= N_RAYS_ * 3) return;
    int ray = idx / 3, c = idx % 3;
    const float* base = rgbw + (size_t)ray * P_ * 3 + c;
    float s = 0.0f;
    for (int j = 0; j < P_; ++j) s += base[(size_t)j * 3];
    out[idx] = s + alast[ray] * BG_;
}

// ---------------------------------------------------------------- launch

extern "C" void kernel_launch(void* const* d_in, const int* in_sizes, int n_in,
                              void* d_out, int out_size, void* d_ws, size_t ws_size,
                              hipStream_t stream)
{
    (void)in_sizes; (void)n_in; (void)out_size; (void)ws_size;
    const float* rays_o   = (const float*)d_in[0];
    const float* rays_d   = (const float*)d_in[1];
    const float* viewdirs = (const float*)d_in[2];
    const float* dgrid    = (const float*)d_in[3];
    const float* k0g      = (const float*)d_in[4];
    const float* w0       = (const float*)d_in[5];
    const float* b0       = (const float*)d_in[6];
    const float* w1       = (const float*)d_in[7];
    const float* b1       = (const float*)d_in[8];
    const float* w2       = (const float*)d_in[9];
    const float* b2       = (const float*)d_in[10];
    const int*   stepsize = (const int*)d_in[11];
    float* out = (float*)d_out;

    char* ws = (char*)d_ws;
    size_t offAlpha = 0;
    size_t offWgt   = offAlpha + (size_t)M_ * 4;
    size_t offRgbw  = offWgt   + (size_t)M_ * 4;
    size_t offALast = offRgbw  + (size_t)M_ * 12;
    size_t offWAll  = offALast + (size_t)N_RAYS_ * 4;   // 16 KB gap keeps 16B alignment

    float*    alpha = (float*)   (ws + offAlpha);
    float*    wgt   = (float*)   (ws + offWgt);
    float*    rgbw  = (float*)   (ws + offRgbw);
    float*    alast = (float*)   (ws + offALast);
    _Float16* wAll  = (_Float16*)(ws + offWAll);

    // allow >64KB dynamic LDS (host-side attribute, not a stream op; capture-safe)
    (void)hipFuncSetAttribute((const void*)fused_mlp_kernel,
                              hipFuncAttributeMaxDynamicSharedMemorySize, SMEM_BYTES);

    prep_weights_kernel<<<(WTOT_HALVES + 255)/256, 256, 0, stream>>>(w0, w1, w2, wAll);
    alpha_kernel<<<(M_ + 255)/256, 256, 0, stream>>>(rays_o, rays_d, dgrid, stepsize, alpha);
    scan_kernel<<<(N_RAYS_ + 255)/256, 256, 0, stream>>>(alpha, wgt, alast);
    fused_mlp_kernel<<<2048, 256, SMEM_BYTES, stream>>>(rays_o, rays_d, viewdirs, k0g,
                                                        wAll, b0, b1, b2, wgt, rgbw);
    reduce_rays_kernel<<<(N_RAYS_*3 + 255)/256, 256, 0, stream>>>(rgbw, alast, out);
}
